// KorniaFeatureExtractor_48799418417489
// MI455X (gfx1250) — compile-verified, matrix operand order checked by reading, not run on hardware
//
#include <hip/hip_runtime.h>

// ---------------------------------------------------------------------------
// CDNA5 / gfx1250 feature extractor.
//   K1 (VALU, memory-bound): per 16x16 cell -> hist/HSV/Sobel/Laplacian feats,
//       per-cell partial sums, and f16 gray plane for the WMMA kernel.
//   K2: per-image reduction of partials -> global RGB mean + HSV mean/std.
//   KB: build WMMA B-fragments (shifted Gabor taps) in exact lane layout.
//   K3 (WMMA): 21x21 Gabor conv as implicit GEMM, v_wmma_f32_16x16x32_f16,
//       84 WMMAs per 16x16 cell; gray tile staged into LDS by the Tensor
//       Data Mover (tensor_load_to_lds + s_wait_tensorcnt) for interior
//       cells, scalar gather fallback at image borders.
// ---------------------------------------------------------------------------

typedef __attribute__((ext_vector_type(16))) _Float16 v16h;
typedef __attribute__((ext_vector_type(8)))  _Float16 v8h;
typedef __attribute__((ext_vector_type(4)))  _Float16 v4h;
typedef __attribute__((ext_vector_type(8)))  float    v8f;
typedef __attribute__((ext_vector_type(4)))  unsigned int v4u;
typedef __attribute__((ext_vector_type(8)))  int      v8i;
typedef __attribute__((ext_vector_type(4)))  int      v4i;

#ifndef __has_builtin
#define __has_builtin(x) 0
#endif
#if __has_builtin(__builtin_amdgcn_tensor_load_to_lds) && \
    __has_builtin(__builtin_amdgcn_s_wait_tensorcnt)
#define USE_TDM 1
#else
#define USE_TDM 0
#endif

#define GDIM   7
#define CELL   16
#define HW     112
#define NPIX   (HW*HW)     // 12544
#define NCELL  49
#define FEAT   1822

// output section offsets (flat per image)
#define OFF_RGB      0      // 3
#define OFF_HIST     3      // 1176 : ((cell*3+c)*8+b)
#define OFF_HSVG     1179   // 6    : hmean,hstd,smean,sstd,vmean,vstd
#define OFF_HSVGRID  1185   // 147  : cell*3+c
#define OFF_GRAD     1332   // 196  : cell*4 + {mean,std,meansq,cosmean}
#define OFF_TEX      1528   // 196  : cell*4 + {mean,std,meansq,absmean}
#define OFF_GAB0     1724   // 49
#define OFF_GAB1     1773   // 49

// ---------------------------------------------------------------------------
// block-wide (256-thread / 8-wave32) sum; result valid in tid 0 only
// ---------------------------------------------------------------------------
__device__ __forceinline__ float blockSum256(float v, float* red) {
  for (int o = 16; o > 0; o >>= 1) v += __shfl_down(v, o, 32);
  const int wave = threadIdx.x >> 5, lane = threadIdx.x & 31;
  __syncthreads();                       // red[] may be in use from prior call
  if (lane == 0) red[wave] = v;
  __syncthreads();
  float s = 0.f;
  if (threadIdx.x < 8) s = red[threadIdx.x];
  if (wave == 0)
    for (int o = 4; o > 0; o >>= 1) s += __shfl_down(s, o, 32);
  return s;
}

// ---------------------------------------------------------------------------
// K1: per (image, cell) feature pass. 256 threads = one thread per pixel.
// ---------------------------------------------------------------------------
__global__ __launch_bounds__(256) void k_cell_features(
    const float* __restrict__ img, _Float16* __restrict__ gray16,
    float* __restrict__ part, float* __restrict__ out)
{
  const int cell = blockIdx.x % NCELL;
  const int bimg = blockIdx.x / NCELL;
  const int gy = cell / GDIM, gxc = cell % GDIM;
  const int y0 = gy * CELL, x0 = gxc * CELL;
  const int tid = threadIdx.x;
  const int ty = tid >> 4, tx = tid & 15;
  const int y = y0 + ty, x = x0 + tx;

  __shared__ float gt[18 * 18];   // gray tile with 1-px clamped halo
  __shared__ int   hist[24];
  __shared__ float red[8];

  const float* base = img + (size_t)bimg * (3 * NPIX);
  const float pr = base[y * HW + x];
  const float pg = base[NPIX + y * HW + x];
  const float pb = base[2 * NPIX + y * HW + x];
  const float gray = 0.299f * pr + 0.587f * pg + 0.114f * pb;

  if (tid < 24) hist[tid] = 0;
  gt[(ty + 1) * 18 + (tx + 1)] = gray;
  gray16[(size_t)bimg * NPIX + y * HW + x] = (_Float16)gray;

  if (tid < 68) {                  // 1-px halo ring, edge-clamped
    int rr, cc;
    if (tid < 18)      { rr = 0;        cc = tid;      }
    else if (tid < 36) { rr = 17;       cc = tid - 18; }
    else if (tid < 52) { rr = tid - 35; cc = 0;        }
    else               { rr = tid - 51; cc = 17;       }
    int yy = y0 - 1 + rr, xx = x0 - 1 + cc;
    int yc = min(max(yy, 0), HW - 1), xc = min(max(xx, 0), HW - 1);
    const float hr = base[yc * HW + xc];
    const float hg = base[NPIX + yc * HW + xc];
    const float hb = base[2 * NPIX + yc * HW + xc];
    gt[rr * 18 + cc] = 0.299f * hr + 0.587f * hg + 0.114f * hb;
  }
  __syncthreads();

  // per-cell, per-channel 8-bin histogram
  {
    const int b0 = (int)floorf(pr * 8.f);
    const int b1 = (int)floorf(pg * 8.f);
    const int b2 = (int)floorf(pb * 8.f);
    if ((unsigned)b0 < 8u) atomicAdd(&hist[b0], 1);
    if ((unsigned)b1 < 8u) atomicAdd(&hist[8 + b1], 1);
    if ((unsigned)b2 < 8u) atomicAdd(&hist[16 + b2], 1);
  }

  // HSV (matches reference formulas)
  const float maxc  = fmaxf(pr, fmaxf(pg, pb));
  const float minc  = fminf(pr, fminf(pg, pb));
  const float vv    = maxc;
  const float delta = maxc - minc;
  const float ssat  = delta / (maxc + 1e-8f);
  const float dsafe = (delta == 0.f) ? 1.f : delta;
  const float rc = (maxc - pr) / dsafe;
  const float gc = (maxc - pg) / dsafe;
  const float bc = (maxc - pb) / dsafe;
  float h = (maxc == pr) ? (bc - gc)
          : ((maxc == pg) ? (2.f + rc - bc) : (4.f + gc - rc));
  h *= (1.f / 6.f);
  h -= floorf(h);                        // python %1.0 semantics
  h *= 6.283185307179586f;

  // Sobel (edge padding comes from clamped halo)
  const int tc = (ty + 1) * 18 + (tx + 1);
  const float tl = gt[tc - 19], tt = gt[tc - 18], tr = gt[tc - 17];
  const float ll = gt[tc - 1],                    rr2 = gt[tc + 1];
  const float bl = gt[tc + 17], bt = gt[tc + 18], br = gt[tc + 19];
  const float gxv = (-tl + tr - 2.f * ll + 2.f * rr2 - bl + br) * 0.125f;
  const float gyv = (-tl - 2.f * tt - tr + bl + 2.f * bt + br) * 0.125f;
  const float mag = sqrtf(gxv * gxv + gyv * gyv + 1e-8f);
  const float ca  = cosf(atan2f(gyv, gxv));

  // Laplacian with reflect padding (reflect only differs at global border,
  // and the reflected index is always inside this tile)
  float lsum = 0.f;
  #pragma unroll
  for (int dyy = -1; dyy <= 1; dyy++)
    #pragma unroll
    for (int dxx = -1; dxx <= 1; dxx++) {
      if (dyy == 0 && dxx == 0) continue;
      int ny = y + dyy; ny = (ny < 0) ? -ny : ((ny > HW - 1) ? 2 * (HW - 1) - ny : ny);
      int nx = x + dxx; nx = (nx < 0) ? -nx : ((nx > HW - 1) ? 2 * (HW - 1) - nx : nx);
      lsum += gt[(ny - (y0 - 1)) * 18 + (nx - (x0 - 1))];
    }
  const float lap = (lsum - 8.f * gray) * (1.f / 16.f);

  // deterministic tree reductions (15 scalars)
  const float rs = blockSum256(pr, red);
  const float gs = blockSum256(pg, red);
  const float bs = blockSum256(pb, red);
  const float hs = blockSum256(h, red);
  const float sS = blockSum256(ssat, red);
  const float vS = blockSum256(vv, red);
  const float hq = blockSum256(h * h, red);
  const float sq = blockSum256(ssat * ssat, red);
  const float vq = blockSum256(vv * vv, red);
  const float ms = blockSum256(mag, red);
  const float mq = blockSum256(mag * mag, red);
  const float cs = blockSum256(ca, red);
  const float ls = blockSum256(lap, red);
  const float lq = blockSum256(lap * lap, red);
  const float la = blockSum256(fabsf(lap), red);

  float* o = out + (size_t)bimg * FEAT;
  if (tid == 0) {
    const float invN = 1.f / 256.f, invM = 1.f / 255.f;
    o[OFF_HSVGRID + cell * 3 + 0] = hs * invN;
    o[OFF_HSVGRID + cell * 3 + 1] = sS * invN;
    o[OFF_HSVGRID + cell * 3 + 2] = vS * invN;
    o[OFF_GRAD + cell * 4 + 0] = ms * invN;
    o[OFF_GRAD + cell * 4 + 1] = sqrtf(fmaxf((mq - ms * ms * invN) * invM, 0.f));
    o[OFF_GRAD + cell * 4 + 2] = mq * invN;
    o[OFF_GRAD + cell * 4 + 3] = cs * invN;
    o[OFF_TEX + cell * 4 + 0] = ls * invN;
    o[OFF_TEX + cell * 4 + 1] = sqrtf(fmaxf((lq - ls * ls * invN) * invM, 0.f));
    o[OFF_TEX + cell * 4 + 2] = lq * invN;
    o[OFF_TEX + cell * 4 + 3] = la * invN;
    float* pp = part + ((size_t)bimg * NCELL + cell) * 9;
    pp[0] = rs; pp[1] = gs; pp[2] = bs;
    pp[3] = hs; pp[4] = sS; pp[5] = vS;
    pp[6] = hq; pp[7] = sq; pp[8] = vq;
  }
  __syncthreads();
  if (tid < 24)  // tid = c*8 + bin ; layout ((cell*3+c)*8+bin)
    o[OFF_HIST + cell * 24 + tid] = (float)hist[tid] * (1.f / 256.f);
}

// ---------------------------------------------------------------------------
// K2: per-image global stats from per-cell partials (deterministic).
// ---------------------------------------------------------------------------
__global__ __launch_bounds__(64) void k_global_stats(
    const float* __restrict__ part, float* __restrict__ out)
{
  const int bimg = blockIdx.x;
  const int tid = threadIdx.x;
  __shared__ float sums[9];
  if (tid < 9) {
    const float* p = part + (size_t)bimg * NCELL * 9 + tid;
    float s = 0.f;
    for (int c = 0; c < NCELL; c++) s += p[c * 9];
    sums[tid] = s;
  }
  __syncthreads();
  if (tid == 0) {
    const float N = (float)NPIX;
    float* o = out + (size_t)bimg * FEAT;
    o[OFF_RGB + 0] = sums[0] / N;
    o[OFF_RGB + 1] = sums[1] / N;
    o[OFF_RGB + 2] = sums[2] / N;
    for (int c = 0; c < 3; c++) {
      const float mean = sums[3 + c] / N;
      const float var  = (sums[6 + c] - sums[3 + c] * sums[3 + c] / N) / (N - 1.f);
      o[OFF_HSVG + 2 * c]     = mean;
      o[OFF_HSVG + 2 * c + 1] = sqrtf(fmaxf(var, 0.f));
    }
  }
}

// ---------------------------------------------------------------------------
// KB: build WMMA B fragments. frag = dy*4 + chunk*2 + gk  (84 frags).
// Lane layout per ISA 16-bit matrix spec: lanes 0-15 -> K {0-7,16-23},
// lanes 16-31 -> K {8-15,24-31}; column N = lane&15; B[k][n] = w[dy][k-n].
// Stored 32 B/lane contiguous so K3 does one v16h (2x b128) load per frag.
// ---------------------------------------------------------------------------
__global__ __launch_bounds__(256) void k_build_bfrag(
    const float* __restrict__ k0, const float* __restrict__ k1,
    _Float16* __restrict__ bf)
{
  const int t = blockIdx.x * 256 + threadIdx.x;
  if (t >= 84 * 32) return;
  const int lane = t & 31;
  const int frag = t >> 5;
  const int gk    = frag & 1;
  const int chunk = (frag >> 1) & 1;
  const int dy    = frag >> 2;
  const float* src = gk ? k1 : k0;
  const int g = (lane >> 4) & 1;
  const int n = lane & 15;
  _Float16* dst = bf + (size_t)frag * 512 + lane * 16;
  #pragma unroll
  for (int i = 0; i < 16; i++) {
    const int K  = 32 * chunk + (i < 8 ? i : i + 8) + 8 * g;
    const int dx = K - n;
    dst[i] = (_Float16)((dx >= 0 && dx <= 20) ? src[dy * 21 + dx] : 0.f);
  }
}

// ---------------------------------------------------------------------------
// K3: Gabor conv as implicit GEMM on v_wmma_f32_16x16x32_f16.
// One wave32 per (image, cell). D tile = 16x16 conv outputs of the cell.
// ---------------------------------------------------------------------------
#define TSTR 48   // LDS row stride in halves (96 B, multiple of 16 B)

__global__ __launch_bounds__(32) void k_gabor_wmma(
    const _Float16* __restrict__ gray16, const _Float16* __restrict__ bf,
    float* __restrict__ out)
{
  const int cell = blockIdx.x % NCELL;
  const int bimg = blockIdx.x / NCELL;
  const int gy = cell / GDIM, gxc = cell % GDIM;
  const int y0 = gy * CELL, x0 = gxc * CELL;
  const int lane = threadIdx.x;

  __shared__ __align__(16) _Float16 tile[36 * TSTR];  // 36-row window

  // B-fragment table is tiny (84 KB) and shared by all 50k workgroups -> L2.
  __builtin_prefetch((const void*)bf, 0, 1);   // -> global_prefetch_b8

  const _Float16* gimg = gray16 + (size_t)bimg * NPIX;

  bool staged = false;
#if USE_TDM
  // Tensor Data Mover: one 2D descriptor stages the whole tile into LDS at a
  // 48-half row stride; tensor_dim0/1 give right/bottom zero-fill past the
  // image edge. Top/left (negative) offsets not expressible -> interior only.
  if (gy > 0 && gxc > 0) {
    const unsigned long long gaddr =
        (unsigned long long)(uintptr_t)(gimg + (y0 - 10) * HW + (x0 - 10));
    const unsigned tdim0 = (unsigned)(122 - x0);   // valid halves from tile x
    const unsigned tdim1 = (unsigned)(122 - y0);   // valid rows from tile y

    v4u g0;                                           // D# group 0 (ISA 8.3)
    g0[0] = 1u;                                       // count=1, user mode
    g0[1] = (unsigned)(uintptr_t)(void*)tile;         // lds_addr (bytes)
    g0[2] = (unsigned)(gaddr & 0xFFFFFFFFu);          // global_addr[31:0]
    g0[3] = (unsigned)((gaddr >> 32) & 0x1FFFFFFu)    // global_addr[56:32]
          | 0x80000000u;                              // type=2 ("image")

    v8i g1;                                           // D# group 1 (ISA 8.4)
    g1[0] = (int)(1u << 16);                          // data_size=1 (2 bytes)
    g1[1] = (int)(tdim0 << 16);                       // tensor_dim0[15:0]
    g1[2] = (int)(tdim1 << 16);                       // dim0 hi=0 | dim1 lo
    g1[3] = (int)(48u << 16);                         // dim1 hi=0 | tile_dim0=48
    g1[4] = 36;                                       // tile_dim1=36, tile_dim2=0
    g1[5] = HW;                                       // tensor_dim0_stride=112
    g1[6] = 0;                                        // stride hi | dim1_stride
    g1[7] = 0;

    v4i z4 = {};
#if __clang_major__ >= 23
    v8i z8 = {};
    __builtin_amdgcn_tensor_load_to_lds(g0, g1, z4, z4, z8, 0);
#else
    __builtin_amdgcn_tensor_load_to_lds(g0, g1, z4, z4, 0);
#endif
    __builtin_amdgcn_s_wait_tensorcnt(0);
    staged = true;
  }
#endif
  if (!staged) {   // border cells (and no-TDM toolchains): scalar gather
    for (int t = lane; t < 36 * 36; t += 32) {
      const int rr = t / 36, cc = t - rr * 36;
      const int yy = y0 - 10 + rr, xx = x0 - 10 + cc;
      _Float16 v = (_Float16)0.f;
      if ((unsigned)yy < (unsigned)HW && (unsigned)xx < (unsigned)HW)
        v = gimg[yy * HW + xx];
      tile[rr * TSTR + cc] = v;
    }
  }
  __syncthreads();

  const int g = lane >> 4;       // K-half selector per ISA A layout
  const int m = lane & 15;       // A row (M)
  const v16h* bfv = (const v16h*)bf;

  v8f acc0 = {};                 // gabor_k0 accumulator (16x16 f32)
  v8f acc1 = {};                 // gabor_k1 accumulator

  for (int dy = 0; dy < 21; dy++) {
    const _Float16* rp = &tile[(m + dy) * TSTR];
    // A chunk 0 : K 0..31  (lanes<16: K{0-7,16-23}; lanes>=16: K{8-15,24-31})
    const v8h lo = *(const v8h*)(rp + 8 * g);        // 16 B aligned
    const v8h hi = *(const v8h*)(rp + 8 * g + 16);   // 16 B aligned
    const v16h a0 = __builtin_shufflevector(lo, hi,
        0, 1, 2, 3, 4, 5, 6, 7, 8, 9, 10, 11, 12, 13, 14, 15);
    // A chunk 1 : K 32..63 ; only K 32..35 live (lanes 0-15, elems 0-3)
    v16h a1 = {};
    if (g == 0) {
      const v4h q = *(const v4h*)(rp + 32);          // 8 B aligned
      a1[0] = q[0]; a1[1] = q[1]; a1[2] = q[2]; a1[3] = q[3];
    }
    const size_t fb = (size_t)(dy * 4) * 32 + lane;
    const v16h b00 = bfv[fb];           // chunk0, gabor0
    const v16h b10 = bfv[fb + 32];      // chunk0, gabor1
    const v16h b01 = bfv[fb + 64];      // chunk1, gabor0
    const v16h b11 = bfv[fb + 96];      // chunk1, gabor1

    acc0 = __builtin_amdgcn_wmma_f32_16x16x32_f16(false, a0, false, b00,
                                                  (short)0, acc0, false, false);
    acc1 = __builtin_amdgcn_wmma_f32_16x16x32_f16(false, a0, false, b10,
                                                  (short)0, acc1, false, false);
    acc0 = __builtin_amdgcn_wmma_f32_16x16x32_f16(false, a1, false, b01,
                                                  (short)0, acc0, false, false);
    acc1 = __builtin_amdgcn_wmma_f32_16x16x32_f16(false, a1, false, b11,
                                                  (short)0, acc1, false, false);
  }

  // cell mean of |response| : 32 lanes x 8 regs cover all 256 outputs
  float s0 = 0.f, s1 = 0.f;
  #pragma unroll
  for (int i = 0; i < 8; i++) { s0 += fabsf(acc0[i]); s1 += fabsf(acc1[i]); }
  for (int o = 16; o > 0; o >>= 1) {
    s0 += __shfl_down(s0, o, 32);
    s1 += __shfl_down(s1, o, 32);
  }
  if (lane == 0) {
    float* o = out + (size_t)bimg * FEAT;
    o[OFF_GAB0 + cell] = s0 * (1.f / 256.f);
    o[OFF_GAB1 + cell] = s1 * (1.f / 256.f);
  }
}

// ---------------------------------------------------------------------------
extern "C" void kernel_launch(void* const* d_in, const int* in_sizes, int n_in,
                              void* d_out, int out_size, void* d_ws, size_t ws_size,
                              hipStream_t stream)
{
  const float* img = (const float*)d_in[0];
  const float* gk0 = (const float*)d_in[1];
  const float* gk1 = (const float*)d_in[2];
  float* out = (float*)d_out;
  const int nimg = in_sizes[0] / (3 * NPIX);

  // workspace: [gray f16 | per-cell partials f32 | B-fragments f16]
  char* w = (char*)d_ws;
  _Float16* gray16 = (_Float16*)w;
  size_t off = ((size_t)nimg * NPIX * sizeof(_Float16) + 255) & ~(size_t)255;
  float* part = (float*)(w + off);
  off += (((size_t)nimg * NCELL * 9 * sizeof(float)) + 255) & ~(size_t)255;
  _Float16* bfrag = (_Float16*)(w + off);   // 84 * 512 halves = 86016 B

  k_build_bfrag<<<dim3((84 * 32 + 255) / 256), dim3(256), 0, stream>>>(gk0, gk1, bfrag);
  k_cell_features<<<dim3(nimg * NCELL), dim3(256), 0, stream>>>(img, gray16, part, out);
  k_global_stats<<<dim3(nimg), dim3(64), 0, stream>>>(part, out);
  k_gabor_wmma<<<dim3(nimg * NCELL), dim3(32), 0, stream>>>(gray16, bfrag, out);
}
